// GlobalEncoder_13116830122156
// MI455X (gfx1250) — matrix operand order repaired and checked by reading, not compiled, not run on hardware
//
#include <hip/hip_runtime.h>
#include <math.h>

// ---------------------------------------------------------------------------
// Problem constants (from reference)
// ---------------------------------------------------------------------------
#define NN   50000   // nodes
#define KNEI 32      // neighbors per node
#define DIM  100     // embedding dim
#define CPAD 112     // DIM padded to 7 * 16 WMMA col tiles
#define CT   (CPAD / 16)           // 7 col tiles

// Kernel 1 tiling
#define NPB1  2                    // nodes per block
#define ROWS1 (NPB1 * KNEI)        // 64 rows of (node, neighbor) pairs
#define KPAD1 128                  // D+1 = 101 padded to 4 * 32
#define KCH1  (KPAD1 / 32)         // 4 K-chunks

// Kernel 2 tiling
#define NPB2  32                   // nodes per block
#define K2    (2 * DIM)            // 200 (concat [x, msg])
#define KPAD2 224                  // 200 padded to 7 * 32
#define KCH2  (KPAD2 / 32)         // 7 K-chunks

typedef __attribute__((ext_vector_type(16))) __bf16 v16bf;
typedef __attribute__((ext_vector_type(8)))  float  v8f;

// ---------------------------------------------------------------------------
// Branchless transcendentals on the hardware TRANS pipe (v_exp_f32/v_rcp_f32).
// tanh(x) = 1 - 2/(e^{2x}+1); exact at +/-inf, no divergent fixup branches.
// ---------------------------------------------------------------------------
#define LOG2E 1.4426950408889634f

__device__ __forceinline__ float fast_tanh(float x) {
  float e = __builtin_amdgcn_exp2f(x * (2.0f * LOG2E));
  return 1.0f - 2.0f * __builtin_amdgcn_rcpf(e + 1.0f);
}

__device__ __forceinline__ float fast_exp(float x) {
  return __builtin_amdgcn_exp2f(x * LOG2E);
}

// ---------------------------------------------------------------------------
// A-fragment loader (LDS row-major -> VGPR) per CDNA5 ISA 7.12.2:
// lanes 0-15 hold row M=lane; half j maps to K = j + 8*(j>=8) + 8*hi.
// Per lane this is two contiguous 8-half (16B) runs -> 2x ds_load_b128.
// Caller guarantees the LDS tile is zero-padded to the full K stride.
// ---------------------------------------------------------------------------
__device__ __forceinline__ v16bf frag_a(const __bf16* p, int row0, int k0,
                                        int stride, int lane) {
  int m  = lane & 15;
  int hi = lane >> 4;
  const __bf16* r = p + (row0 + m) * stride + k0 + 8 * hi;
  v16bf a;
#pragma unroll
  for (int j = 0; j < 8; ++j) a[j] = r[j];           // K = k0+8hi .. +7
#pragma unroll
  for (int j = 0; j < 8; ++j) a[8 + j] = r[16 + j];  // K = k0+16+8hi .. +7
  return a;
}

// ---------------------------------------------------------------------------
// B-fragment loader from pre-swizzled LDS: each (kchunk, coltile, lane) owns
// 16 contiguous bf16 (32 B, 32B-aligned) -> one v16bf load (2x ds_load_b128).
// ---------------------------------------------------------------------------
__device__ __forceinline__ v16bf frag_b_sw(const __bf16* swz, int kc, int ct,
                                           int lane) {
  const v16bf* pv = reinterpret_cast<const v16bf*>(
      swz + (((kc * CT) + ct) * 32 + lane) * 16);
  return *pv;
}

// ---------------------------------------------------------------------------
// Kernel 1: attention scores (WMMA bf16) + softmax + weighted message -> ws
// ---------------------------------------------------------------------------
__global__ __launch_bounds__(256)
void ge_attn_msg_kernel(const int* __restrict__ nei,
                        const float* __restrict__ wei,
                        const float* __restrict__ s_vec,
                        const float* __restrict__ emb,
                        const float* __restrict__ W1_w,
                        const float* __restrict__ W1_b,
                        const float* __restrict__ q1_w,
                        float* __restrict__ msg) {
  __shared__ __attribute__((aligned(32))) __bf16 featA[ROWS1][KPAD1]; // 16384 B
  __shared__ __attribute__((aligned(8)))  __bf16 hneiS[ROWS1][DIM];   // 12800 B
  // W1 pre-swizzled into B-fragment layout: [KCH1][CT][lane][16]     // 28672 B
  __shared__ __attribute__((aligned(32))) __bf16 w1Bf[KCH1 * CT * 32 * 16];
  __shared__ float  q1f[CPAD], b1f[CPAD];
  __shared__ float  scoreS[ROWS1], attS[ROWS1];
  __shared__ float  svecS[NPB1][DIM];
  __shared__ int    nidS[ROWS1];
  __shared__ float  wS[ROWS1];

  const int tid   = threadIdx.x;
  const int node0 = blockIdx.x * NPB1;

  // ---- stage A: ids, weights, s_vec, q1/b1, score init ----
  for (int i = tid; i < ROWS1; i += 256) {
    int n   = node0 + (i >> 5);
    int nid = nei[n * KNEI + (i & 31)];
    nidS[i]   = nid;
    wS[i]     = wei[n * KNEI + (i & 31)];
    scoreS[i] = 0.0f;
    __builtin_prefetch(&emb[nid * DIM], 0, 0);  // global_prefetch of gather row
  }
  for (int i = tid; i < NPB1 * DIM; i += 256)
    svecS[i / DIM][i % DIM] = s_vec[(node0 + i / DIM) * DIM + (i % DIM)];
  for (int i = tid; i < CPAD; i += 256) {
    q1f[i] = (i < DIM) ? q1_w[i] : 0.0f;
    b1f[i] = (i < DIM) ? W1_b[i] : 0.0f;
  }

  // ---- stage A2: W1 -> swizzled B-fragment layout (zero padded) ----
  for (int i = tid; i < KCH1 * CT * 32 * 16; i += 256) {
    int j    = i & 15;
    int lane = (i >> 4) & 31;
    int t    = i >> 9;                 // kc * CT + ct
    int ct   = t % CT;
    int kc   = t / CT;
    int k = kc * 32 + ((lane >> 4) << 3) + (j & 7) + ((j >> 3) << 4);
    int d = ct * 16 + (lane & 15);
    w1Bf[i] = (k <= DIM && d < DIM) ? (__bf16)W1_w[k * DIM + d] : (__bf16)0.0f;
  }
  __syncthreads();

  // ---- stage B: gather h_nei, build zero-padded feat tile ----
  for (int i = tid; i < ROWS1 * KPAD1; i += 256) {
    int r = i >> 7, d = i & (KPAD1 - 1);
    float v;
    if (d < DIM) {
      float h = emb[nidS[r] * DIM + d];
      hneiS[r][d] = (__bf16)h;
      v = h * svecS[r >> 5][d];
    } else if (d == DIM) {
      v = wS[r];
    } else {
      v = 0.0f;
    }
    featA[r][d] = (__bf16)v;
  }
  __syncthreads();

  // ---- stage C: score GEMM, 16x16 tiles, 4 chained K=32 WMMAs ----
  const int wave = tid >> 5;
  const int lane = tid & 31;
  const int RT = ROWS1 / 16;   // 4 row tiles
  for (int job = wave; job < RT * CT; job += 8) {
    int rt = job / CT, ct = job % CT;
    int row0 = rt * 16, col0 = ct * 16;
    v8f acc = {};
#pragma unroll
    for (int kc = 0; kc < KCH1; ++kc) {
      v16bf a = frag_a(&featA[0][0], row0, kc * 32, KPAD1, lane);
      v16bf b = frag_b_sw(w1Bf, kc, ct, lane);
      acc = __builtin_amdgcn_wmma_f32_16x16x32_bf16(
          false, a, false, b, (short)0, acc, false, false);
    }
    // epilogue: score[row] += tanh(acc + b1[col]) * q1[col]  (branchless)
    int nc = lane & 15, hi = lane >> 4;
    int col = col0 + nc;
    float q = q1f[col], bb = b1f[col];
#pragma unroll
    for (int v = 0; v < 8; ++v) {
      float t = fast_tanh(acc[v] + bb);
      atomicAdd(&scoreS[row0 + v + 8 * hi], t * q);
    }
  }
  __syncthreads();

  // ---- stage D: softmax over K=32 (one wave per node, exact wave32 fit) ----
  if (tid < ROWS1) {
    float s = scoreS[tid];
    float m = s;
    for (int off = 16; off > 0; off >>= 1) m = fmaxf(m, __shfl_xor(m, off, 32));
    float e = fast_exp(s - m);
    float sum = e;
    for (int off = 16; off > 0; off >>= 1) sum += __shfl_xor(sum, off, 32);
    attS[tid] = e * __builtin_amdgcn_rcpf(sum);
  }
  __syncthreads();

  // ---- stage E: msg[n,d] = sum_k att * h_nei, two cols per thread ----
  if (tid < NPB1 * (DIM / 2)) {
    int n = tid / (DIM / 2);
    int d = (tid % (DIM / 2)) * 2;
    float a0 = 0.0f, a1 = 0.0f;
#pragma unroll
    for (int k = 0; k < KNEI; ++k) {
      float w = attS[n * KNEI + k];
      a0 += w * (float)hneiS[n * KNEI + k][d];
      a1 += w * (float)hneiS[n * KNEI + k][d + 1];
    }
    float2* mp = reinterpret_cast<float2*>(&msg[(node0 + n) * DIM + d]);
    *mp = make_float2(a0, a1);
  }
}

// ---------------------------------------------------------------------------
// Kernel 2: two ReLU layers x = relu([x, msg] @ W2 + b2) with WMMA bf16
// ---------------------------------------------------------------------------
__global__ __launch_bounds__(256)
void ge_layers_kernel(const int* __restrict__ nodes,
                      const float* __restrict__ emb,
                      const float* __restrict__ W2_w,
                      const float* __restrict__ W2_b,
                      const float* __restrict__ msg,
                      float* __restrict__ out) {
  __shared__ __attribute__((aligned(32))) __bf16 catA[NPB2][KPAD2]; // 14336 B
  // W2 pre-swizzled into B-fragment layout: [KCH2][CT][lane][16]   // 50176 B
  __shared__ __attribute__((aligned(32))) __bf16 w2Bf[KCH2 * CT * 32 * 16];
  __shared__ float b2f[CPAD];                                       //   448 B

  const int tid   = threadIdx.x;
  const int node0 = blockIdx.x * NPB2;

  for (int i = tid; i < CPAD; i += 256)
    b2f[i] = (i < DIM) ? W2_b[i] : 0.0f;
  for (int i = tid; i < KCH2 * CT * 32 * 16; i += 256) {
    int j    = i & 15;
    int lane = (i >> 4) & 31;
    int t    = i >> 9;
    int ct   = t % CT;
    int kc   = t / CT;
    int k = kc * 32 + ((lane >> 4) << 3) + (j & 7) + ((j >> 3) << 4);
    int d = ct * 16 + (lane & 15);
    w2Bf[i] = (k < K2 && d < DIM) ? (__bf16)W2_w[k * DIM + d] : (__bf16)0.0f;
  }
  for (int i = tid; i < NPB2 * KPAD2; i += 256) {
    int r = i / KPAD2, c = i % KPAD2;
    int n = node0 + r;
    float v = 0.0f;
    if (n < NN && c < K2)
      v = (c < DIM) ? emb[nodes[n] * DIM + c] : msg[n * DIM + (c - DIM)];
    catA[r][c] = (__bf16)v;
  }
  __syncthreads();

  const int wave = tid >> 5;
  const int lane = tid & 31;
  const int RT = NPB2 / 16;    // 2 row tiles
  const int nc = lane & 15, hi = lane >> 4;

  float dres[2][8];            // per-wave job results (<= 2 jobs per wave)

  for (int layer = 0; layer < 2; ++layer) {
    int nj = 0;
    for (int job = wave; job < RT * CT; job += 8, ++nj) {
      int rt = job / CT, ct = job % CT;
      int row0 = rt * 16;
      v8f acc = {};
#pragma unroll
      for (int kc = 0; kc < KCH2; ++kc) {
        v16bf a = frag_a(&catA[0][0], row0, kc * 32, KPAD2, lane);
        v16bf b = frag_b_sw(w2Bf, kc, ct, lane);
        acc = __builtin_amdgcn_wmma_f32_16x16x32_bf16(
            false, a, false, b, (short)0, acc, false, false);
      }
      float bb = b2f[ct * 16 + nc];
#pragma unroll
      for (int v = 0; v < 8; ++v)
        dres[nj][v] = fmaxf(acc[v] + bb, 0.0f);
    }
    __syncthreads();   // all reads of catA done before write-back

    if (layer == 0) {
      nj = 0;
      for (int job = wave; job < RT * CT; job += 8, ++nj) {
        int row0 = (job / CT) * 16;
        int col  = (job % CT) * 16 + nc;
        if (col < DIM)
#pragma unroll
          for (int v = 0; v < 8; ++v)
            catA[row0 + v + 8 * hi][col] = (__bf16)dres[nj][v];
      }
      __syncthreads();
    } else {
      nj = 0;
      for (int job = wave; job < RT * CT; job += 8, ++nj) {
        int row0 = (job / CT) * 16;
        int col  = (job % CT) * 16 + nc;
        if (col < DIM)
#pragma unroll
          for (int v = 0; v < 8; ++v) {
            int n = node0 + row0 + v + 8 * hi;
            if (n < NN) out[n * DIM + col] = dres[nj][v];
          }
      }
    }
  }
}

// ---------------------------------------------------------------------------
// Launch
// ---------------------------------------------------------------------------
extern "C" void kernel_launch(void* const* d_in, const int* in_sizes, int n_in,
                              void* d_out, int out_size, void* d_ws, size_t ws_size,
                              hipStream_t stream) {
  const int*   nodes = (const int*)d_in[0];
  const int*   nei   = (const int*)d_in[1];
  const float* wei   = (const float*)d_in[2];
  const float* s_vec = (const float*)d_in[3];
  const float* emb   = (const float*)d_in[4];
  const float* W1_w  = (const float*)d_in[5];
  const float* W1_b  = (const float*)d_in[6];
  const float* q1_w  = (const float*)d_in[7];
  const float* W2_w  = (const float*)d_in[8];
  const float* W2_b  = (const float*)d_in[9];
  float*       out   = (float*)d_out;
  float*       msg   = (float*)d_ws;   // N * D floats = 20 MB scratch

  (void)in_sizes; (void)n_in; (void)out_size; (void)ws_size;

  dim3 blk(256);
  dim3 g1(NN / NPB1);                   // 25000 blocks, 2 nodes each
  dim3 g2((NN + NPB2 - 1) / NPB2);      // 1563 blocks, 32 nodes each

  ge_attn_msg_kernel<<<g1, blk, 0, stream>>>(nei, wei, s_vec, emb,
                                             W1_w, W1_b, q1_w, msg);
  ge_layers_kernel<<<g2, blk, 0, stream>>>(nodes, emb, W2_w, W2_b, msg, out);
}
